// PCPP_8065948582195
// MI455X (gfx1250) — matrix operand order repaired and checked by reading, not compiled, not run on hardware
//
#include <hip/hip_runtime.h>

typedef _Float16 h16;
typedef __attribute__((ext_vector_type(16))) _Float16 v16h;
typedef __attribute__((ext_vector_type(8)))  _Float16 v8h;
typedef __attribute__((ext_vector_type(8)))  float    v8f;

#define DEVINL __device__ __forceinline__

// ---------------- WMMA fragment loaders (CDNA5 16x16x32 f16 layouts) ----------------
// A (16x32 f16, W rows): lanes 0-15 row m, halves 0..7 -> K 0..7, 8..15 -> K 16..23;
//                        lanes 16-31 row m, halves 0..7 -> K 8..15, 8..15 -> K 24..31.
DEVINL v16h frag_a_load(const h16* __restrict__ p, int ld) {
  int lane = (int)(threadIdx.x & 31u);
  int m  = lane & 15;
  int h0 = (lane >> 4) & 1;
  const h16* base = p + (size_t)m * ld + (h0 << 3);
  v8h lo = *(const v8h*)(base);
  v8h hi = *(const v8h*)(base + 16);
  v16h r;
#pragma unroll
  for (int j = 0; j < 8; ++j) { r[j] = lo[j]; r[j + 8] = hi[j]; }
  return r;
}

// B (32x16 f16, act columns, stored [point][channel]): lanes 0-15 col n, K 0..7, 16..23;
// lanes 16-31 col n, K 8..15, 24..31.
DEVINL v16h frag_b_load(const h16* __restrict__ p, int ld) {
  int lane = (int)(threadIdx.x & 31u);
  int n  = lane & 15;
  int k0 = (lane >> 4) << 3;
  const h16* base = p + (size_t)n * ld + k0;
  v8h lo = *(const v8h*)(base);
  v8h hi = *(const v8h*)(base + 16);
  v16h r;
#pragma unroll
  for (int j = 0; j < 8; ++j) { r[j] = lo[j]; r[j + 8] = hi[j]; }
  return r;
}

// One MLP layer: out[co, P] = relu(W[co,cin] @ actin[cin, P] + bias).
// actin in LDS as [point][channel] (stride ld_in halves). Each wave owns 16 columns.
// Register-blocked 2 row-tiles per iteration (co is always a multiple of 32) so each
// B-fragment LDS read feeds two v_wmma issues.
// If gout != nullptr: write f32 to gout[row*gstride + col]; else f16 to actout[col*ld_out + row].
DEVINL void mlp_layer(const h16* __restrict__ actin, int ld_in, int cin,
                      const h16* __restrict__ W, const float* __restrict__ bias, int co,
                      h16* actout, int ld_out, float* gout, int gstride)
{
  int wave = (int)(threadIdx.x >> 5);
  int lane = (int)(threadIdx.x & 31u);
  int col0 = wave << 4;
  int n = lane & 15;
  int mbase = (lane >> 4) << 3;   // D layout: vgpr r -> M = r + 8*(lane>=16)
  int nk = cin >> 5;
  const h16* bcol = actin + (size_t)col0 * ld_in;
  for (int r0 = 0; r0 < co; r0 += 32) {
    const h16* wrow0 = W + (size_t)r0 * cin;
    const h16* wrow1 = wrow0 + (size_t)16 * cin;
    v8f acc0 = {0.f,0.f,0.f,0.f,0.f,0.f,0.f,0.f};
    v8f acc1 = {0.f,0.f,0.f,0.f,0.f,0.f,0.f,0.f};
    for (int kc = 0; kc < nk; ++kc) {
      v16h bm = frag_b_load(bcol + kc * 32, ld_in);
      v16h a0 = frag_a_load(wrow0 + kc * 32, cin);
      v16h a1 = frag_a_load(wrow1 + kc * 32, cin);
      acc0 = __builtin_amdgcn_wmma_f32_16x16x32_f16(false, a0, false, bm, (short)0, acc0, false, false);
      acc1 = __builtin_amdgcn_wmma_f32_16x16x32_f16(false, a1, false, bm, (short)0, acc1, false, false);
    }
#pragma unroll
    for (int r = 0; r < 8; ++r) {
      int row0 = r0 + mbase + r;
      int row1 = row0 + 16;
      float v0 = acc0[r] + bias[row0];
      float v1 = acc1[r] + bias[row1];
      v0 = v0 > 0.f ? v0 : 0.f;
      v1 = v1 > 0.f ? v1 : 0.f;
      if (gout) {
        gout[(size_t)row0 * gstride + (col0 + n)] = v0;
        gout[(size_t)row1 * gstride + (col0 + n)] = v1;
      } else {
        actout[(size_t)(col0 + n) * ld_out + row0] = (h16)v0;
        actout[(size_t)(col0 + n) * ld_out + row1] = (h16)v1;
      }
    }
  }
}

// ---------------- fused SA branch: gather -> 3x(GEMM+bias+relu) -> maxpool over K -------------
template<int PTILE>
__global__ void __launch_bounds__(256)
sa_branch_kernel(const float* __restrict__ pts,   // [B][N][3]
                 const float* __restrict__ feat,  // [B][D][N] or null
                 const float* __restrict__ newx,  // [B][S][3]
                 const int*   __restrict__ gidx,  // [B][S][K]
                 const h16* __restrict__ W0, const float* __restrict__ bias0, int cip0, int co0,
                 const h16* __restrict__ W1, const float* __restrict__ bias1, int co1,
                 const h16* __restrict__ W2, const float* __restrict__ bias2, int co2,
                 float* __restrict__ out, int ctot, int co_off,
                 int N, int D, int S, int K, int ldA, int ldB)
{
  extern __shared__ __attribute__((aligned(16))) h16 smh[];
  h16* bufA = smh;
  h16* bufB = smh + (size_t)PTILE * ldA;
  int b = blockIdx.y;
  int G = PTILE / K;
  int s0 = blockIdx.x * G;
  int tid = (int)threadIdx.x, nt = (int)blockDim.x;

  // stage grouped features: channels = [feat(D), centered xyz(3), zero pad]
  for (int idx = tid; idx < PTILE * cip0; idx += nt) {
    int p = idx % PTILE, c = idx / PTILE;
    int s = s0 + p / K, k = p % K;
    int g = gidx[((size_t)b * S + s) * K + k];
    float v = 0.f;
    if (c < D) v = feat[((size_t)b * D + c) * N + g];
    else if (c < D + 3) {
      int cc = c - D;
      v = pts[((size_t)b * N + g) * 3 + cc] - newx[((size_t)b * S + s) * 3 + cc];
    }
    bufA[(size_t)p * ldA + c] = (h16)v;
  }
  __syncthreads();
  mlp_layer(bufA, ldA, cip0, W0, bias0, co0, bufB, ldB, nullptr, 0);
  __syncthreads();
  mlp_layer(bufB, ldB, co0,  W1, bias1, co1, bufA, ldA, nullptr, 0);
  __syncthreads();
  mlp_layer(bufA, ldA, co1,  W2, bias2, co2, bufB, ldB, nullptr, 0);
  __syncthreads();
  // max pool over K, write [B][ctot][S] slice
  for (int idx = tid; idx < G * co2; idx += nt) {
    int q = idx / co2, c = idx % co2;
    float m = -1e30f;
    for (int k = 0; k < K; ++k) {
      float v = (float)bufB[(size_t)(q * K + k) * ldB + c];
      m = v > m ? v : m;
    }
    out[((size_t)b * ctot + co_off + c) * S + (s0 + q)] = m;
  }
}

// ---------------- generic dense chain with (optional) 3-NN interp staging --------------------
template<int PTILE>
__global__ void __launch_bounds__(256)
fp_mlp_kernel(const float* __restrict__ feat1, int C1,      // [B][C1][N]
              const float* __restrict__ feat2, int C2, int M, // [B][C2][M]; M==1 -> broadcast
              const int* __restrict__ kni, const float* __restrict__ knw, // [B][N][3]
              const h16* __restrict__ W0, const float* __restrict__ bias0, int cip0, int co0,
              const h16* __restrict__ W1, const float* __restrict__ bias1, int co1,
              const h16* __restrict__ W2, const float* __restrict__ bias2, int co2,
              float* __restrict__ out, int N, int ldA, int ldB)
{
  extern __shared__ __attribute__((aligned(16))) h16 smh[];
  h16* bufA = smh;
  h16* bufB = smh + (size_t)PTILE * ldA;
  int b = blockIdx.y;
  int n0 = blockIdx.x * PTILE;
  int tid = (int)threadIdx.x, nt = (int)blockDim.x;

  for (int idx = tid; idx < PTILE * cip0; idx += nt) {
    int p = idx % PTILE, c = idx / PTILE;
    int n = n0 + p;
    float v = 0.f;
    if (c < C1) v = feat1[((size_t)b * C1 + c) * N + n];
    else if (c < C1 + C2) {
      int c2 = c - C1;
      const float* f2 = feat2 + ((size_t)b * C2 + c2) * M;
      if (M == 1) v = f2[0];
      else {
        const int*   ip = kni + ((size_t)b * N + n) * 3;
        const float* wp = knw + ((size_t)b * N + n) * 3;
        v = wp[0] * f2[ip[0]] + wp[1] * f2[ip[1]] + wp[2] * f2[ip[2]];
      }
    }
    bufA[(size_t)p * ldA + c] = (h16)v;
  }
  __syncthreads();
  if (co1 == 0) {
    mlp_layer(bufA, ldA, cip0, W0, bias0, co0, nullptr, 0,
              out + (size_t)b * co0 * N + n0, N);
  } else if (co2 == 0) {
    mlp_layer(bufA, ldA, cip0, W0, bias0, co0, bufB, ldB, nullptr, 0);
    __syncthreads();
    mlp_layer(bufB, ldB, co0, W1, bias1, co1, nullptr, 0,
              out + (size_t)b * co1 * N + n0, N);
  } else {
    mlp_layer(bufA, ldA, cip0, W0, bias0, co0, bufB, ldB, nullptr, 0);
    __syncthreads();
    mlp_layer(bufB, ldB, co0, W1, bias1, co1, bufA, ldA, nullptr, 0);
    __syncthreads();
    mlp_layer(bufA, ldA, co1, W2, bias2, co2, nullptr, 0,
              out + (size_t)b * co2 * N + n0, N);
  }
}

// ---------------- FPS: one workgroup per batch, dist[] in LDS, block argmax ------------------
__global__ void fps_kernel(const float* __restrict__ pts, int N, int npoint,
                           int* __restrict__ outIdx)
{
  extern __shared__ float fsm[];
  float* dist = fsm;                       // N
  float* rval = fsm + N;                   // blockDim
  int*   ridx = (int*)(rval + blockDim.x); // blockDim
  int b = blockIdx.x;
  int tid = (int)threadIdx.x, nt = (int)blockDim.x;
  const float* P = pts + (size_t)b * N * 3;
  for (int i = tid; i < N; i += nt) dist[i] = 1e10f;
  __syncthreads();
  int far = 0;
  for (int it = 0; it < npoint; ++it) {
    if (tid == 0) outIdx[(size_t)b * npoint + it] = far;
    float cx = P[3 * far], cy = P[3 * far + 1], cz = P[3 * far + 2];
    float best = -1.f; int bi = 0;
    for (int i = tid; i < N; i += nt) {
      float dx = P[3*i] - cx, dy = P[3*i+1] - cy, dz = P[3*i+2] - cz;
      float d = dx*dx + dy*dy + dz*dz;
      float dd = dist[i]; dd = d < dd ? d : dd; dist[i] = dd;
      if (dd > best) { best = dd; bi = i; }
    }
    rval[tid] = best; ridx[tid] = bi;
    __syncthreads();
    for (int s = nt >> 1; s > 0; s >>= 1) {
      if (tid < s) {
        float ov = rval[tid + s]; int oi = ridx[tid + s];
        if (ov > rval[tid] || (ov == rval[tid] && oi < ridx[tid])) { rval[tid] = ov; ridx[tid] = oi; }
      }
      __syncthreads();
    }
    far = ridx[0];
    __syncthreads();
  }
}

__global__ void ballquery_kernel(const float* __restrict__ pts, const float* __restrict__ newx,
                                 int N, int S, int K, float r2, int* __restrict__ gidx)
{
  int t = (int)(blockIdx.x * blockDim.x + threadIdx.x);
  int b = blockIdx.y;
  if (t >= S) return;
  const float* q = newx + ((size_t)b * S + t) * 3;
  float qx = q[0], qy = q[1], qz = q[2];
  const float* P = pts + (size_t)b * N * 3;
  int* g = gidx + ((size_t)b * S + t) * K;
  int cnt = 0, first = 0;
  for (int n = 0; n < N && cnt < K; ++n) {
    float dx = P[3*n] - qx, dy = P[3*n+1] - qy, dz = P[3*n+2] - qz;
    float d = dx*dx + dy*dy + dz*dz;
    if (d <= r2) { if (cnt == 0) first = n; g[cnt++] = n; }
  }
  for (int j = cnt; j < K; ++j) g[j] = first;
}

__global__ void knn3_kernel(const float* __restrict__ xf, const float* __restrict__ xc,
                            int N, int M, int* __restrict__ ki, float* __restrict__ kw)
{
  int t = (int)(blockIdx.x * blockDim.x + threadIdx.x);
  int b = blockIdx.y;
  if (t >= N) return;
  const float* q = xf + ((size_t)b * N + t) * 3;
  const float* C = xc + (size_t)b * M * 3;
  float qx = q[0], qy = q[1], qz = q[2];
  float d0 = 1e30f, d1 = 1e30f, d2v = 1e30f; int i0 = 0, i1 = 0, i2 = 0;
  for (int m = 0; m < M; ++m) {
    float dx = C[3*m] - qx, dy = C[3*m+1] - qy, dz = C[3*m+2] - qz;
    float d = dx*dx + dy*dy + dz*dz;
    if (d < d0)      { d2v = d1; i2 = i1; d1 = d0; i1 = i0; d0 = d; i0 = m; }
    else if (d < d1) { d2v = d1; i2 = i1; d1 = d; i1 = m; }
    else if (d < d2v){ d2v = d; i2 = m; }
  }
  float w0 = 1.f/(d0+1e-8f), w1 = 1.f/(d1+1e-8f), w2 = 1.f/(d2v+1e-8f);
  float s = w0 + w1 + w2;
  size_t o = ((size_t)b * N + t) * 3;
  ki[o] = i0; ki[o+1] = i1; ki[o+2] = i2;
  kw[o] = w0/s; kw[o+1] = w1/s; kw[o+2] = w2/s;
}

__global__ void xyz_transpose_kernel(const float* __restrict__ in, float* __restrict__ out, int N)
{ // [B][3][N] -> [B][N][3]
  int t = (int)(blockIdx.x * blockDim.x + threadIdx.x);
  int b = blockIdx.y;
  if (t >= N) return;
  out[((size_t)b*N + t)*3 + 0] = in[((size_t)b*3 + 0)*N + t];
  out[((size_t)b*N + t)*3 + 1] = in[((size_t)b*3 + 1)*N + t];
  out[((size_t)b*N + t)*3 + 2] = in[((size_t)b*3 + 2)*N + t];
}

__global__ void gather_newx_kernel(const float* __restrict__ pts, const int* __restrict__ idx,
                                   int N, int S, float* __restrict__ out)
{
  int t = (int)(blockIdx.x * blockDim.x + threadIdx.x);
  int b = blockIdx.y;
  if (t >= S) return;
  int g = idx[(size_t)b * S + t];
  out[((size_t)b*S + t)*3 + 0] = pts[((size_t)b*N + g)*3 + 0];
  out[((size_t)b*S + t)*3 + 1] = pts[((size_t)b*N + g)*3 + 1];
  out[((size_t)b*S + t)*3 + 2] = pts[((size_t)b*N + g)*3 + 2];
}

__global__ void sa4_concat_kernel(const float* __restrict__ x3, const float* __restrict__ l3p,
                                  float* __restrict__ out, int Np, int D)
{ // out [B][D+3][Np], channels: xyz first, then feats (sa_group_all order)
  int t = (int)(blockIdx.x * blockDim.x + threadIdx.x);
  int b = blockIdx.y;
  int tot = (D + 3) * Np;
  if (t >= tot) return;
  int c = t / Np, n = t % Np;
  float v = (c < 3) ? x3[((size_t)b*Np + n)*3 + c] : l3p[((size_t)b*D + (c-3))*Np + n];
  out[(size_t)b * (D + 3) * Np + t] = v;
}

__global__ void maxpool_kernel(const float* __restrict__ in, float* __restrict__ out, int C, int Np)
{
  int c = (int)(blockIdx.x * blockDim.x + threadIdx.x);
  int b = blockIdx.y;
  if (c >= C) return;
  const float* p = in + ((size_t)b * C + c) * Np;
  float m = p[0];
  for (int n = 1; n < Np; ++n) m = p[n] > m ? p[n] : m;
  out[(size_t)b * C + c] = m;
}

__global__ void wconv_kernel(const float* __restrict__ W, h16* __restrict__ out,
                             int Co, int Ci, int Cip)
{
  long idx = (long)blockIdx.x * blockDim.x + threadIdx.x;
  long tot = (long)Co * Cip;
  if (idx >= tot) return;
  int r = (int)(idx / Cip), c = (int)(idx % Cip);
  out[idx] = (c < Ci) ? (h16)W[(size_t)r * Ci + c] : (h16)0.f;
}

// ------------------------------------ host orchestration ------------------------------------
static inline int imax2(int a, int b) { return a > b ? a : b; }

extern "C" void kernel_launch(void* const* d_in, const int* in_sizes, int n_in,
                              void* d_out, int out_size, void* d_ws, size_t ws_size,
                              hipStream_t stream) {
  (void)in_sizes; (void)n_in; (void)out_size; (void)ws_size;
  const int B = 4, N1 = 8192, S1 = 1024, S2 = 512, S3 = 128;

  (void)hipFuncSetAttribute(reinterpret_cast<const void*>(&sa_branch_kernel<64>),
                            hipFuncAttributeMaxDynamicSharedMemorySize, 160*1024);
  (void)hipFuncSetAttribute(reinterpret_cast<const void*>(&sa_branch_kernel<128>),
                            hipFuncAttributeMaxDynamicSharedMemorySize, 240*1024);
  (void)hipFuncSetAttribute(reinterpret_cast<const void*>(&fp_mlp_kernel<64>),
                            hipFuncAttributeMaxDynamicSharedMemorySize, 250*1024);

  // ---- workspace allocator ----
  size_t off = 0;
  auto alloc = [&](size_t bytes) -> void* {
    off = (off + 255) & ~(size_t)255;
    void* p = (void*)((char*)d_ws + off);
    off += bytes;
    return p;
  };

  // ---- collect layers (d_in order) and allocate padded f16 weights ----
  struct LayerW { const float* W; const float* b; int ci, co, cip; h16* wh; };
  LayerW L[37]; int nl = 0; int pi = 2;
  auto addLayer = [&](int ci, int co) {
    LayerW lw;
    lw.W = (const float*)d_in[pi++]; lw.b = (const float*)d_in[pi++];
    lw.ci = ci; lw.co = co; lw.cip = (ci + 31) & ~31;
    lw.wh = (h16*)alloc((size_t)co * lw.cip * sizeof(h16));
    L[nl++] = lw;
  };
  static const int sa1c[3][4] = {{3,32,32,64},{3,64,64,128},{3,64,96,128}};
  static const int sa2c[3][4] = {{323,64,64,128},{323,128,128,256},{323,128,128,256}};
  static const int sa3c[3][4] = {{643,64,64,128},{643,128,128,256},{643,128,128,256}};
  for (int br = 0; br < 3; ++br) { int ci = sa1c[br][0]; for (int l = 0; l < 3; ++l) { addLayer(ci, sa1c[br][l+1]); ci = sa1c[br][l+1]; } }
  for (int br = 0; br < 3; ++br) { int ci = sa2c[br][0]; for (int l = 0; l < 3; ++l) { addLayer(ci, sa2c[br][l+1]); ci = sa2c[br][l+1]; } }
  for (int br = 0; br < 3; ++br) { int ci = sa3c[br][0]; for (int l = 0; l < 3; ++l) { addLayer(ci, sa3c[br][l+1]); ci = sa3c[br][l+1]; } }
  { int ci = 643; int ch[3] = {256,512,1024}; for (int l = 0; l < 3; ++l) { addLayer(ci, ch[l]); ci = ch[l]; } }  // sa4 -> 27..29
  { int ci = 1664; int ch[2] = {256,256}; for (int l = 0; l < 2; ++l) { addLayer(ci, ch[l]); ci = ch[l]; } }      // fp3 -> 30..31
  { int ci = 896;  int ch[2] = {256,128}; for (int l = 0; l < 2; ++l) { addLayer(ci, ch[l]); ci = ch[l]; } }      // fp2 -> 32..33
  { int ci = 448;  int ch[3] = {128,128,128}; for (int l = 0; l < 3; ++l) { addLayer(ci, ch[l]); ci = ch[l]; } }  // fp1 -> 34..36

  for (int i = 0; i < nl; ++i) {
    long tot = (long)L[i].co * L[i].cip;
    wconv_kernel<<<dim3((unsigned)((tot + 255) / 256)), 256, 0, stream>>>(L[i].W, L[i].wh, L[i].co, L[i].ci, L[i].cip);
  }

  // ---- activation / index buffers ----
  float* x1  = (float*)alloc((size_t)B*N1*3*4);
  int* fidx1 = (int*)alloc((size_t)B*S1*4);
  int* fidx2 = (int*)alloc((size_t)B*S2*4);
  int* fidx3 = (int*)alloc((size_t)B*S3*4);
  float* nx1 = (float*)alloc((size_t)B*S1*3*4);
  float* nx2 = (float*)alloc((size_t)B*S2*3*4);
  float* nx3 = (float*)alloc((size_t)B*S3*3*4);
  int* gi    = (int*)alloc((size_t)B*S1*32*4);      // max B*S*K over all branches
  float* l1p = (float*)alloc((size_t)B*320*S1*4);
  float* l2p = (float*)alloc((size_t)B*640*S2*4);
  float* l3p = (float*)alloc((size_t)B*640*S3*4);
  float* sa4cat = (float*)alloc((size_t)B*643*S3*4);
  float* sa4h1  = (float*)alloc((size_t)B*256*S3*4);
  float* sa4h2  = (float*)alloc((size_t)B*512*S3*4);
  float* sa4h3  = (float*)alloc((size_t)B*1024*S3*4);
  float* l4p    = (float*)alloc((size_t)B*1024*4);
  float* fp3o   = (float*)alloc((size_t)B*256*S3*4);
  float* fp2o   = (float*)alloc((size_t)B*128*S2*4);
  int*   ki     = (int*)alloc((size_t)B*S1*3*4);
  float* kw     = (float*)alloc((size_t)B*S1*3*4);

  const float* xyz_in = (const float*)d_in[0];
  xyz_transpose_kernel<<<dim3((N1+255)/256, B), 256, 0, stream>>>(xyz_in, x1, N1);

  // ---- SA stage runner ----
  auto run_sa = [&](const float* pts, const float* feat, int D, int N, int S,
                    const float* radii, const int* Ks, int base,
                    int* fidx, float* nx, float* outp, int ctot) {
    size_t fsm = (size_t)N*4 + 256*8;
    fps_kernel<<<B, 256, fsm, stream>>>(pts, N, S, fidx);
    gather_newx_kernel<<<dim3((S+255)/256, B), 256, 0, stream>>>(pts, fidx, N, S, nx);
    int co_off = 0;
    for (int br = 0; br < 3; ++br) {
      int K = Ks[br]; float r2 = radii[br]*radii[br];
      ballquery_kernel<<<dim3((S+127)/128, B), 128, 0, stream>>>(pts, nx, N, S, K, r2, gi);
      LayerW& l0 = L[base + br*3 + 0];
      LayerW& l1 = L[base + br*3 + 1];
      LayerW& l2 = L[base + br*3 + 2];
      int PT  = (K > 64) ? 128 : 64;
      int ldA = imax2(l0.cip, l1.co);
      int ldB = imax2(l0.co, l2.co);
      size_t smem = (size_t)PT * (ldA + ldB) * sizeof(h16);
      int G = PT / K;
      dim3 grid(S / G, B);
      if (PT == 64) {
        sa_branch_kernel<64><<<grid, 128, smem, stream>>>(pts, feat, nx, gi,
          l0.wh, l0.b, l0.cip, l0.co, l1.wh, l1.b, l1.co, l2.wh, l2.b, l2.co,
          outp, ctot, co_off, N, D, S, K, ldA, ldB);
      } else {
        sa_branch_kernel<128><<<grid, 256, smem, stream>>>(pts, feat, nx, gi,
          l0.wh, l0.b, l0.cip, l0.co, l1.wh, l1.b, l1.co, l2.wh, l2.b, l2.co,
          outp, ctot, co_off, N, D, S, K, ldA, ldB);
      }
      co_off += l2.co;
    }
  };

  { static const float r[3] = {0.01f,0.02f,0.03f}; static const int k[3] = {16,16,32};
    run_sa(x1,  nullptr, 0,   N1, S1, r, k, 0,  fidx1, nx1, l1p, 320); }
  { static const float r[3] = {0.02f,0.03f,0.04f}; static const int k[3] = {32,32,64};
    run_sa(nx1, l1p,     320, S1, S2, r, k, 9,  fidx2, nx2, l2p, 640); }
  { static const float r[3] = {0.03f,0.04f,0.05f}; static const int k[3] = {64,64,128};
    run_sa(nx2, l2p,     640, S2, S3, r, k, 18, fidx3, nx3, l3p, 640); }

  // ---- SA4 (group_all): concat xyz-first, 3 dense layers, global max-pool ----
  sa4_concat_kernel<<<dim3((643*S3 + 255)/256, B), 256, 0, stream>>>(nx3, l3p, sa4cat, S3, 640);
  {
    LayerW& l = L[27];
    size_t smem = (size_t)64 * l.cip * sizeof(h16);
    fp_mlp_kernel<64><<<dim3(S3/64, B), 128, smem, stream>>>(sa4cat, 643, nullptr, 0, 0,
      nullptr, nullptr, l.wh, l.b, l.cip, l.co,
      nullptr, nullptr, 0, nullptr, nullptr, 0, sa4h1, S3, l.cip, 0);
  }
  {
    LayerW& l = L[28];
    size_t smem = (size_t)64 * l.cip * sizeof(h16);
    fp_mlp_kernel<64><<<dim3(S3/64, B), 128, smem, stream>>>(sa4h1, 256, nullptr, 0, 0,
      nullptr, nullptr, l.wh, l.b, l.cip, l.co,
      nullptr, nullptr, 0, nullptr, nullptr, 0, sa4h2, S3, l.cip, 0);
  }
  {
    LayerW& l = L[29];
    size_t smem = (size_t)64 * l.cip * sizeof(h16);
    fp_mlp_kernel<64><<<dim3(S3/64, B), 128, smem, stream>>>(sa4h2, 512, nullptr, 0, 0,
      nullptr, nullptr, l.wh, l.b, l.cip, l.co,
      nullptr, nullptr, 0, nullptr, nullptr, 0, sa4h3, S3, l.cip, 0);
  }
  maxpool_kernel<<<dim3(4, B), 256, 0, stream>>>(sa4h3, l4p, 1024, S3);

  // ---- FP3: broadcast interp (M==1), layers 1664->256->256, N=128 ----
  {
    LayerW& f0 = L[30]; LayerW& f1 = L[31];
    int ldA = f0.cip, ldB = f0.co;
    size_t smem = (size_t)64 * (ldA + ldB) * sizeof(h16);
    fp_mlp_kernel<64><<<dim3(S3/64, B), 128, smem, stream>>>(l3p, 640, l4p, 1024, 1,
      nullptr, nullptr, f0.wh, f0.b, f0.cip, f0.co,
      f1.wh, f1.b, f1.co, nullptr, nullptr, 0, fp3o, S3, ldA, ldB);
  }
  // ---- FP2: 3-NN from nx3(128) onto nx2(512), layers 896->256->128 ----
  knn3_kernel<<<dim3((S2+127)/128, B), 128, 0, stream>>>(nx2, nx3, S2, S3, ki, kw);
  {
    LayerW& f0 = L[32]; LayerW& f1 = L[33];
    int ldA = f0.cip, ldB = f0.co;
    size_t smem = (size_t)64 * (ldA + ldB) * sizeof(h16);
    fp_mlp_kernel<64><<<dim3(S2/64, B), 128, smem, stream>>>(l2p, 640, fp3o, 256, S3,
      ki, kw, f0.wh, f0.b, f0.cip, f0.co,
      f1.wh, f1.b, f1.co, nullptr, nullptr, 0, fp2o, S2, ldA, ldB);
  }
  // ---- FP1: 3-NN from nx2(512) onto nx1(1024), layers 448->128->128->128 -> d_out ----
  knn3_kernel<<<dim3((S1+127)/128, B), 128, 0, stream>>>(nx1, nx2, S1, S2, ki, kw);
  {
    LayerW& f0 = L[34]; LayerW& f1 = L[35]; LayerW& f2 = L[36];
    int ldA = imax2(f0.cip, f1.co);
    int ldB = imax2(f0.co, f2.co);
    size_t smem = (size_t)64 * (ldA + ldB) * sizeof(h16);
    fp_mlp_kernel<64><<<dim3(S1/64, B), 128, smem, stream>>>(l1p, 320, fp2o, 128, S2,
      ki, kw, f0.wh, f0.b, f0.cip, f0.co,
      f1.wh, f1.b, f1.co, f2.wh, f2.b, f2.co, (float*)d_out, S1, ldA, ldB);
  }
}
